// RelationMessagePassingBase_45973329937216
// MI455X (gfx1250) — matrix operand order repaired and checked by reading, not compiled, not run on hardware
//
#include <hip/hip_runtime.h>

typedef __attribute__((ext_vector_type(16))) __bf16 v16bf;
typedef __attribute__((ext_vector_type(8)))  __bf16 v8bf;
typedef __attribute__((ext_vector_type(4)))  __bf16 v4bf;
typedef __attribute__((ext_vector_type(8)))  float  v8f;

// Branchless fast mish: mish(x) = x * tanh(softplus(x))
//   t = e^x ;  tanh(softplus(x)) = t(t+2) / (t(t+2)+2)
// Clamp exp input at 20: there the ratio is 1 - O(1e-17), matching mish(x)->x.
__device__ __forceinline__ float mishf(float x) {
    float t = __expf(fminf(x, 20.0f));
    float u = t * (t + 2.0f);
    return x * (u * __builtin_amdgcn_rcpf(u + 2.0f));
}

// fp32 -> bf16 weight pre-conversion (weights are tiny: 1.5MB total fp32)
__global__ void cvt_bf16_kernel(const float* __restrict__ src,
                                __bf16* __restrict__ dst, int n4) {
    int i = blockIdx.x * 256 + threadIdx.x;
    if (i < n4) {
        float4 v = ((const float4*)src)[i];
        v4bf b;
        b[0] = (__bf16)v.x; b[1] = (__bf16)v.y;
        b[2] = (__bf16)v.z; b[3] = (__bf16)v.w;
        ((v4bf*)dst)[i] = b;
    }
}

// One block: 128 facts x full d = 64*ARITY, 8 waves, each wave owns a private
// 16-row slab of the LDS tile (no cross-wave traffic after the gather).
template <int ARITY>
__global__ __launch_bounds__(256)
void rel_msg_kernel(const float*  __restrict__ emb,   // [N_NODES, 64]
                    const int*    __restrict__ idx,   // [G, F*ARITY]
                    const __bf16* __restrict__ wiB,   // [G, d, d] bf16
                    const float*  __restrict__ bi,    // [G, d]
                    const __bf16* __restrict__ woB,   // [G, d, d] bf16
                    const float*  __restrict__ bo,    // [G, d]
                    float*        __restrict__ out,   // group-offset base, rows of 64
                    int F)
{
    constexpr int D   = 64 * ARITY;
    constexpr int PAD = 8;                // bf16 elements of row padding (16B)
    constexpr int LS  = D + PAD;          // LDS row stride in bf16 elements
    constexpr int TM  = 128;              // facts per block
    constexpr int KT  = D / 32;           // K tiles (K=32 per WMMA)
    constexpr int NT  = D / 16;           // N tiles per wave

    extern __shared__ __bf16 lds[];       // TM * LS bf16

    const int tid  = threadIdx.x;
    const int g    = blockIdx.y;
    const int f0   = blockIdx.x * TM;
    const int lane = tid & 31;
    const int ln   = lane & 15;           // N (or M) index within 16
    const int half = lane >> 4;           // 0: lanes 0-15, 1: lanes 16-31
    const int wv   = tid >> 5;            // wave id: row block

    const int* idxg = idx + (size_t)g * (size_t)F * ARITY;

    // ---------------- gather: fp32 embeddings -> bf16 LDS tile ----------------
    constexpr int CHUNKS = TM * ARITY * 16;   // float4 chunks
    for (int c = tid; c < CHUNKS; c += 256) {
        const int s  = c >> 4;                 // node slot within tile
        const int q  = c & 15;                 // float4 index within 64-float row
        const int fl = s / ARITY;              // local fact row
        const int j  = s - fl * ARITY;
        const int f  = f0 + fl;
        const int node = (f < F) ? idxg[(size_t)f * ARITY + j] : 0;
        const float4 v = ((const float4*)(emb + (size_t)node * 64))[q];
        v4bf b;
        b[0] = (__bf16)v.x; b[1] = (__bf16)v.y;
        b[2] = (__bf16)v.z; b[3] = (__bf16)v.w;
        *(v4bf*)(lds + (size_t)fl * LS + j * 64 + q * 4) = b;
    }
    __syncthreads();

    __bf16* xw = lds + (size_t)wv * 16 * LS;   // this wave's 16-row slab

#pragma unroll
    for (int phase = 0; phase < 2; ++phase) {
        const __bf16* W  = phase ? woB : wiB;
        const float*  Bp = phase ? bo  : bi;

        // Cache all A fragments for this wave's 16 rows in registers.
        // A fragment, contiguous k order matched by B: lane ln = row M; elem i -> K = k*32+half*16+i.
        v16bf afrag[KT];
#pragma unroll
        for (int k = 0; k < KT; ++k) {
            const __bf16* p = xw + ln * LS + k * 32 + half * 16;
            v8bf lo = *(const v8bf*)p;
            v8bf hi = *(const v8bf*)(p + 8);
            afrag[k] = __builtin_shufflevector(lo, hi,
                         0,1,2,3,4,5,6,7,8,9,10,11,12,13,14,15);
        }

        for (int n = 0; n < NT; ++n) {
            v8f acc = {};
            // B 32x16 fragment, same contiguous k order as A:
            // lane ln = column N; elem i -> K = k*32 + half*16 + i.
            // B[k][n] = W[o = n*16+ln][k]  (x @ W^T): 16 contiguous bf16 = 32B.
            const __bf16* wrow = W + ((size_t)g * D + (size_t)(n * 16 + ln)) * D
                                   + half * 16;
#pragma unroll
            for (int k = 0; k < KT; ++k) {
                v16bf bf = *(const v16bf*)(wrow + k * 32);
                acc = __builtin_amdgcn_wmma_f32_16x16x32_bf16(
                          false, afrag[k], false, bf, (short)0, acc, false, false);
            }

            // C/D layout: lane ln = column N = n*16+ln; VGPR v = row M = v + half*8.
            const int   col  = n * 16 + ln;
            const float bval = Bp[(size_t)g * D + col];

            if (phase == 0) {
                // h = mish(x@wi^T + bi) -> bf16, in place over this wave's slab
                // (safe: A fragments already register-cached).
#pragma unroll
                for (int v = 0; v < 8; ++v) {
                    float t = mishf(acc[v] + bval);
                    xw[(v + half * 8) * LS + col] = (__bf16)t;
                }
            } else {
                // out = x (fp32 re-gather from L2-resident table) + h@wo^T + bo
                const int j = col >> 6;
                const int e = col & 63;
#pragma unroll
                for (int v = 0; v < 8; ++v) {
                    const int f = f0 + wv * 16 + v + half * 8;
                    if (f < F) {
                        const int node = idxg[(size_t)f * ARITY + j];
                        const float xv = emb[(size_t)node * 64 + e];
                        const size_t orow = ((size_t)g * F + (size_t)f) * ARITY + j;
                        // streaming store: don't evict the embedding table from L2
                        __builtin_nontemporal_store(acc[v] + bval + xv,
                                                    out + orow * 64 + e);
                    }
                }
            }
        }
        // phase0 LDS stores -> phase1 LDS loads are same-wave DS ops (in order).
    }
}

extern "C" void kernel_launch(void* const* d_in, const int* in_sizes, int n_in,
                              void* d_out, int out_size, void* d_ws, size_t ws_size,
                              hipStream_t stream) {
    (void)n_in; (void)out_size; (void)ws_size;

    const float* emb = (const float*)d_in[0];
    const int AR[4] = {1, 2, 3, 4};
    const int NG[4] = {2, 3, 2, 1};

    // total message rows = sum over groups of G*F*a == sum of idx element counts
    size_t total_rows = 0;
    int Fs[4];
    for (int i = 0; i < 4; ++i) {
        const int base = 1 + i * 5;
        Fs[i] = in_sizes[base] / (NG[i] * AR[i]);
        total_rows += (size_t)in_sizes[base];
    }
    float* out_msgs = (float*)d_out;
    int*   out_idx  = (int*)(out_msgs + total_rows * 64);

    // ---- pre-convert all weight matrices fp32 -> bf16 into d_ws (~786KB) ----
    __bf16* wsB = (__bf16*)d_ws;
    __bf16* wiBp[4]; __bf16* woBp[4];
    size_t woff = 0;
    for (int i = 0; i < 4; ++i) {
        const int base = 1 + i * 5;
        const int d = 64 * AR[i];
        const size_t welems = (size_t)NG[i] * d * d;   // per matrix set
        wiBp[i] = wsB + woff;
        {
            int n4 = (int)(welems / 4);
            cvt_bf16_kernel<<<(n4 + 255) / 256, 256, 0, stream>>>(
                (const float*)d_in[base + 1], wiBp[i], n4);
        }
        woff += welems;
        woBp[i] = wsB + woff;
        {
            int n4 = (int)(welems / 4);
            cvt_bf16_kernel<<<(n4 + 255) / 256, 256, 0, stream>>>(
                (const float*)d_in[base + 3], woBp[i], n4);
        }
        woff += welems;
    }

    // ---- main message-passing GEMMs, one launch per arity group ----
    size_t row_off = 0;   // E-rows into message region
    size_t idx_off = 0;   // ints into index region
    for (int i = 0; i < 4; ++i) {
        const int base = 1 + i * 5;
        const int*   idx = (const int*)  d_in[base + 0];
        const float* bi  = (const float*)d_in[base + 2];
        const float* bo  = (const float*)d_in[base + 4];
        const int F = Fs[i];

        dim3 grid((F + 127) / 128, NG[i]);
        const size_t ldsB = (size_t)128 * (64 * AR[i] + 8) * 2;  // bf16 tile
        float* outp = out_msgs + row_off * 64;

        switch (AR[i]) {
        case 1: rel_msg_kernel<1><<<grid, 256, ldsB, stream>>>(emb, idx, wiBp[i], bi, woBp[i], bo, outp, F); break;
        case 2: rel_msg_kernel<2><<<grid, 256, ldsB, stream>>>(emb, idx, wiBp[i], bi, woBp[i], bo, outp, F); break;
        case 3: rel_msg_kernel<3><<<grid, 256, ldsB, stream>>>(emb, idx, wiBp[i], bi, woBp[i], bo, outp, F); break;
        case 4: rel_msg_kernel<4><<<grid, 256, ldsB, stream>>>(emb, idx, wiBp[i], bi, woBp[i], bo, outp, F); break;
        }

        // output_indices is exactly the concatenation of the idx inputs
        hipMemcpyAsync(out_idx + idx_off, idx, (size_t)in_sizes[base] * sizeof(int),
                       hipMemcpyDeviceToDevice, stream);

        row_off += (size_t)in_sizes[base];
        idx_off += (size_t)in_sizes[base];
    }
}
